// GCN_dev_11149735101022
// MI455X (gfx1250) — matrix operand order repaired and checked by reading, not compiled, not run on hardware
//
#include <hip/hip_runtime.h>
#include <hip/hip_bf16.h>

// ---------------------------------------------------------------------------
// GCN (jraph GraphConvolution x2) for MI455X / gfx1250, wave32 + WMMA bf16.
// Round 2: native f32 atomics (global_atomic_add_f32) + b128-vectorized
// edge gather / GEMV loads. WMMA MLP unchanged (confirmed good codegen).
// ---------------------------------------------------------------------------

typedef __attribute__((ext_vector_type(16))) __bf16 bf16x16;
typedef __attribute__((ext_vector_type(8)))  __bf16 bf16x8;
typedef __attribute__((ext_vector_type(8)))  float  f32x8;
typedef __attribute__((ext_vector_type(4)))  float  f32x4;

#define EMBED_DIM     128
#define WG_THREADS    256      // 8 waves of 32
#define WAVES_PER_WG  8
#define TILE_ROWS     128      // 16 rows per wave

// Native hardware f32 atomic add (global_atomic_add_f32). The aggregation
// buffers live in coarse-grained device memory (d_ws), so this is safe.
__device__ __forceinline__ void atomic_add_f32(float* p, float v) {
  unsafeAtomicAdd(p, v);
}

// ---- prep: transpose W1/W2 to bf16, column(n)-major so B fragments are
//      contiguous 32B runs per lane -----------------------------------------
__global__ __launch_bounds__(WG_THREADS)
void gcn_prep_weights(const float* __restrict__ W1, const float* __restrict__ W2,
                      __bf16* __restrict__ W1T, __bf16* __restrict__ W2T) {
  int idx = blockIdx.x * WG_THREADS + threadIdx.x;
  if (idx < EMBED_DIM * EMBED_DIM) {
    int k = idx >> 7, n = idx & 127;
    W1T[n * EMBED_DIM + k] = (__bf16)W1[k * EMBED_DIM + n];
    W2T[n * EMBED_DIM + k] = (__bf16)W2[k * EMBED_DIM + n];
  }
}

// ---- degrees (self edge contributes the initial 1.0) -----------------------
__global__ __launch_bounds__(WG_THREADS)
void gcn_init_deg(float* __restrict__ deg_s, float* __restrict__ deg_r, int N) {
  int i = blockIdx.x * WG_THREADS + threadIdx.x;
  if (i < N) { deg_s[i] = 1.0f; deg_r[i] = 1.0f; }
}

__global__ __launch_bounds__(WG_THREADS)
void gcn_accum_deg(const int* __restrict__ senders, const int* __restrict__ receivers,
                   float* __restrict__ deg_s, float* __restrict__ deg_r, int E) {
  int e = blockIdx.x * WG_THREADS + threadIdx.x;
  if (e < E) {
    atomic_add_f32(&deg_s[senders[e]],   1.0f);
    atomic_add_f32(&deg_r[receivers[e]], 1.0f);
  }
}

__global__ __launch_bounds__(WG_THREADS)
void gcn_rsqrt_deg(const float* __restrict__ deg_s, const float* __restrict__ deg_r,
                   float* __restrict__ rs_s, float* __restrict__ rs_r, int N) {
  int i = blockIdx.x * WG_THREADS + threadIdx.x;
  if (i < N) {
    rs_s[i] = rsqrtf(fmaxf(deg_s[i], 1.0f));
    rs_r[i] = rsqrtf(fmaxf(deg_r[i], 1.0f));
  }
}

// ---- fused MLP: h = lrelu(lrelu(X@W1+b1)@W2+b2) * rsqrt(deg_s) -------------
// One 128-row tile per workgroup; each wave owns 16 rows and runs
// 2 layers x 4 K-steps x 8 N-tiles = 64 v_wmma_f32_16x16x32_bf16.
// B fragments stream from global: W1T/W2T are 64KB total and stay resident in
// the 192MB L2, so re-reads across waves are effectively free.
__global__ __launch_bounds__(WG_THREADS)
void gcn_mlp_wmma(const int*   __restrict__ node_ids,
                  const float* __restrict__ embed,
                  const __bf16* __restrict__ W1T,
                  const __bf16* __restrict__ W2T,
                  const float* __restrict__ b1,
                  const float* __restrict__ b2,
                  const float* __restrict__ rs_s,
                  float* __restrict__ h_scaled,
                  float* __restrict__ agg,
                  int N) {
  // per-wave 16x128 bf16 staging tile in A-matrix-friendly row-major layout
  __shared__ __align__(64) __bf16 lds_h1[WAVES_PER_WG][16][EMBED_DIM];

  const int lane = threadIdx.x & 31;
  const int wave = threadIdx.x >> 5;
  const int hgrp = lane >> 4;   // lane half-group: selects K run / D row group
  const int nl   = lane & 15;   // A-row (M) index, also B/C/D column index
  const int rowBase = blockIdx.x * TILE_ROWS + wave * 16;

  // A-fragment source row (clamped; EXEC stays all-ones for WMMA)
  int arow = rowBase + nl;
  arow = (arow < N) ? arow : (N - 1);
  const float* xr0 = embed + (size_t)node_ids[arow] * EMBED_DIM;

  // ---------------- layer 1 ----------------
  f32x8 acc[8];
#pragma unroll
  for (int nt = 0; nt < 8; ++nt) acc[nt] = (f32x8){0,0,0,0,0,0,0,0};

#pragma unroll
  for (int kt = 0; kt < 4; ++kt) {
    const float* xr = xr0 + kt * 32;
    f32x8 xlo = *(const f32x8*)(xr + hgrp * 16);       // K = kt*32 + hgrp*16 ..
    f32x8 xhi = *(const f32x8*)(xr + hgrp * 16 + 8);   // K = kt*32 + hgrp*16 + 8 ..
    bf16x16 afrag;
#pragma unroll
    for (int i = 0; i < 8; ++i) { afrag[i] = (__bf16)xlo[i]; afrag[i + 8] = (__bf16)xhi[i]; }
#pragma unroll
    for (int nt = 0; nt < 8; ++nt) {
      // B lane layout: col = nl, 16 consecutive K values starting at kt*32+hgrp*16
      const bf16x16 bfrag =
          *(const bf16x16*)(W1T + (size_t)(nt * 16 + nl) * EMBED_DIM + kt * 32 + hgrp * 16);
      acc[nt] = __builtin_amdgcn_wmma_f32_16x16x32_bf16(
          false, afrag, false, bfrag, (short)0, acc[nt], false, false);
    }
  }

  // bias + leaky_relu(0.01) then stage as bf16 (D layout -> row-major tile)
#pragma unroll
  for (int nt = 0; nt < 8; ++nt) {
    float bias = b1[nt * 16 + nl];
#pragma unroll
    for (int r = 0; r < 8; ++r) {
      float v = acc[nt][r] + bias;
      v = (v >= 0.0f) ? v : 0.01f * v;
      lds_h1[wave][hgrp * 8 + r][nt * 16 + nl] = (__bf16)v;
    }
  }
  __syncthreads();

  // ---------------- layer 2 ----------------
  f32x8 acc2[8];
#pragma unroll
  for (int nt = 0; nt < 8; ++nt) acc2[nt] = (f32x8){0,0,0,0,0,0,0,0};

#pragma unroll
  for (int kt = 0; kt < 4; ++kt) {
    const __bf16* hr = &lds_h1[wave][nl][kt * 32];
    bf16x8 alo = *(const bf16x8*)(hr + hgrp * 16);
    bf16x8 ahi = *(const bf16x8*)(hr + hgrp * 16 + 8);
    bf16x16 afrag = __builtin_shufflevector(alo, ahi,
        0,1,2,3,4,5,6,7,8,9,10,11,12,13,14,15);
#pragma unroll
    for (int nt = 0; nt < 8; ++nt) {
      const bf16x16 bfrag =
          *(const bf16x16*)(W2T + (size_t)(nt * 16 + nl) * EMBED_DIM + kt * 32 + hgrp * 16);
      acc2[nt] = __builtin_amdgcn_wmma_f32_16x16x32_bf16(
          false, afrag, false, bfrag, (short)0, acc2[nt], false, false);
    }
  }

  // bias + leaky_relu, scale by rsqrt(deg_s); write h_scaled and the self-edge
  // initialization of the aggregation buffer.
  float rs[8]; int rowv[8];
#pragma unroll
  for (int r = 0; r < 8; ++r) {
    int row = rowBase + hgrp * 8 + r;
    rowv[r] = row;
    rs[r] = (row < N) ? rs_s[row] : 0.0f;
  }
#pragma unroll
  for (int nt = 0; nt < 8; ++nt) {
    float bias = b2[nt * 16 + nl];
    int col = nt * 16 + nl;
#pragma unroll
    for (int r = 0; r < 8; ++r) {
      float v = acc2[nt][r] + bias;
      v = (v >= 0.0f) ? v : 0.01f * v;
      v *= rs[r];
      if (rowv[r] < N) {
        size_t o = (size_t)rowv[r] * EMBED_DIM + col;
        h_scaled[o] = v;
        agg[o] = v;     // self edge
      }
    }
  }
}

// ---- edge scatter-add of 128-float rows: one wave per edge -----------------
// Lane l owns columns [4l, 4l+4): one global_load_b128 gather per lane,
// then 4 native global_atomic_add_f32 per lane.
__global__ __launch_bounds__(WG_THREADS)
void gcn_edge_agg_vec(const int* __restrict__ senders, const int* __restrict__ receivers,
                      const float* __restrict__ h_scaled, float* __restrict__ agg, int E) {
  const int lane = threadIdx.x & 31;
  const int wave = threadIdx.x >> 5;
  int e = blockIdx.x * WAVES_PER_WG + wave;
  if (e >= E) return;
  int s = senders[e], r = receivers[e];
  const f32x4 v = *(const f32x4*)(h_scaled + (size_t)s * EMBED_DIM + lane * 4);
  float* ar = agg + (size_t)r * EMBED_DIM + lane * 4;
#pragma unroll
  for (int c = 0; c < 4; ++c) {
    atomic_add_f32(ar + c, v[c]);
  }
}

// ---- GEMV: y = (agg * rsqrt(deg_r)) @ W3 + b3, then pre-scale for agg #2 ----
__global__ __launch_bounds__(WG_THREADS)
void gcn_gemv(const float* __restrict__ agg, const float* __restrict__ W3,
              const float* __restrict__ b3, const float* __restrict__ rs_r,
              const float* __restrict__ rs_s,
              float* __restrict__ y_scaled, float* __restrict__ agg2, int N) {
  const int lane = threadIdx.x & 31;
  const int wave = threadIdx.x >> 5;
  int i = blockIdx.x * WAVES_PER_WG + wave;
  if (i >= N) return;
  const f32x4 rv = *(const f32x4*)(agg + (size_t)i * EMBED_DIM + lane * 4);
  const f32x4 wv = *(const f32x4*)(W3 + lane * 4);
  float p = rv[0] * wv[0] + rv[1] * wv[1] + rv[2] * wv[2] + rv[3] * wv[3];
#pragma unroll
  for (int off = 16; off > 0; off >>= 1) p += __shfl_xor(p, off, 32);
  if (lane == 0) {
    float y  = p * rs_r[i] + b3[0];     // h3 = agg*rs_r ; y = h3@W3 + b3
    float ys = y * rs_s[i];             // pre-normalize for second aggregation
    y_scaled[i] = ys;
    agg2[i] = ys;                       // self edge
  }
}

__global__ __launch_bounds__(WG_THREADS)
void gcn_edge_agg_scalar(const int* __restrict__ senders, const int* __restrict__ receivers,
                         const float* __restrict__ y_scaled, float* __restrict__ agg2, int E) {
  int e = blockIdx.x * WG_THREADS + threadIdx.x;
  if (e < E) atomic_add_f32(&agg2[receivers[e]], y_scaled[senders[e]]);
}

// ---- sigmoid + reference masking (N_CLASSES == 1) ---------------------------
__global__ __launch_bounds__(WG_THREADS)
void gcn_finalize(const float* __restrict__ agg2, const float* __restrict__ rs_r,
                  float* __restrict__ out, int N) {
  int i = blockIdx.x * WG_THREADS + threadIdx.x;
  if (i < N) {
    float v = agg2[i] * rs_r[i];
    v = 1.0f / (1.0f + __expf(-v));
    const int col = 0;                    // N_CLASSES == 1
    if (i == 0) v = 0.0f;                 // h[0,:] = 0
    if (col == 0) v = 0.0f;               // h[:,0] = 0
    if (i == 0 && col == 0) v = 1.0f;     // h[0,0] = 1
    out[i] = v;
  }
}

// ---------------------------------------------------------------------------
extern "C" void kernel_launch(void* const* d_in, const int* in_sizes, int n_in,
                              void* d_out, int out_size, void* d_ws, size_t ws_size,
                              hipStream_t stream) {
  const int*   node_ids  = (const int*)  d_in[0];
  const int*   senders   = (const int*)  d_in[1];
  const int*   receivers = (const int*)  d_in[2];
  const float* embed     = (const float*)d_in[3];
  const float* W1        = (const float*)d_in[4];
  const float* b1        = (const float*)d_in[5];
  const float* W2        = (const float*)d_in[6];
  const float* b2        = (const float*)d_in[7];
  const float* W3        = (const float*)d_in[8];
  const float* b3        = (const float*)d_in[9];

  const int N = in_sizes[0];   // 50000
  const int E = in_sizes[1];   // 800000

  // workspace carve-up (256B aligned slices)
  char* ws = (char*)d_ws;
  size_t off = 0;
  auto carve = [&](size_t bytes) -> char* {
    char* p = ws + off;
    off += (bytes + 255) & ~(size_t)255;
    return p;
  };
  __bf16* W1T      = (__bf16*)carve(EMBED_DIM * EMBED_DIM * sizeof(__bf16));
  __bf16* W2T      = (__bf16*)carve(EMBED_DIM * EMBED_DIM * sizeof(__bf16));
  float*  h_scaled = (float*) carve((size_t)N * EMBED_DIM * sizeof(float));
  float*  agg      = (float*) carve((size_t)N * EMBED_DIM * sizeof(float));
  float*  deg_s    = (float*) carve((size_t)N * sizeof(float));
  float*  deg_r    = (float*) carve((size_t)N * sizeof(float));
  float*  rs_s     = (float*) carve((size_t)N * sizeof(float));
  float*  rs_r     = (float*) carve((size_t)N * sizeof(float));
  float*  y_scaled = (float*) carve((size_t)N * sizeof(float));
  float*  agg2     = (float*) carve((size_t)N * sizeof(float));
  (void)ws_size; (void)n_in;

  const int nBlk = (N + WG_THREADS - 1) / WG_THREADS;
  const int eBlk = (E + WG_THREADS - 1) / WG_THREADS;

  gcn_prep_weights<<<(EMBED_DIM * EMBED_DIM + WG_THREADS - 1) / WG_THREADS,
                     WG_THREADS, 0, stream>>>(W1, W2, W1T, W2T);
  gcn_init_deg<<<nBlk, WG_THREADS, 0, stream>>>(deg_s, deg_r, N);
  gcn_accum_deg<<<eBlk, WG_THREADS, 0, stream>>>(senders, receivers, deg_s, deg_r, E);
  gcn_rsqrt_deg<<<nBlk, WG_THREADS, 0, stream>>>(deg_s, deg_r, rs_s, rs_r, N);

  gcn_mlp_wmma<<<(N + TILE_ROWS - 1) / TILE_ROWS, WG_THREADS, 0, stream>>>(
      node_ids, embed, W1T, W2T, b1, b2, rs_s, h_scaled, agg, N);

  gcn_edge_agg_vec<<<(E + WAVES_PER_WG - 1) / WAVES_PER_WG, WG_THREADS, 0, stream>>>(
      senders, receivers, h_scaled, agg, E);

  gcn_gemv<<<(N + WAVES_PER_WG - 1) / WAVES_PER_WG, WG_THREADS, 0, stream>>>(
      agg, W3, b3, rs_r, rs_s, y_scaled, agg2, N);

  gcn_edge_agg_scalar<<<eBlk, WG_THREADS, 0, stream>>>(senders, receivers, y_scaled, agg2, E);

  gcn_finalize<<<nBlk, WG_THREADS, 0, stream>>>(agg2, rs_r, (float*)d_out, N);
}
